// KNNEmbedding_45191645889144
// MI455X (gfx1250) — matrix operand order repaired and checked by reading, not compiled, not run on hardware
//
#include <hip/hip_runtime.h>
#include <hip/hip_bf16.h>
#include <cstdint>
#include <cstddef>

typedef __attribute__((ext_vector_type(16))) _Float16 v16h;
typedef __attribute__((ext_vector_type(8)))  _Float16 v8h;
typedef __attribute__((ext_vector_type(8)))  float    v8f;

#define B_    16
#define P_    2048
#define D_    16
#define KNN_  16
#define DM_   512              // d_model
#define CIN_  512              // K*2*D
#define COUT_ 1024             // 2*d_model
#define NPTS_ (B_*P_)          // 32768
#define EPS_  1e-5f

// ---------------------------------------------------------------------------
// Kernel 1: masked per-(batch, channel) mean/std over P. 512 blocks, one per
// (b, c) with c in [0,32): c<16 -> crd channel, c>=16 -> ftr channel.
// ---------------------------------------------------------------------------
__global__ __launch_bounds__(256) void k_stats(const float* __restrict__ x,
                                               const float* __restrict__ features,
                                               const unsigned char* __restrict__ mask,
                                               float* __restrict__ meanv,
                                               float* __restrict__ stdv)
{
    int bc = blockIdx.x;           // 0..511
    int b  = bc >> 5;
    int c  = bc & 31;
    int d  = c & 15;
    bool isftr = (c >= 16);
    bool ff = features[b * D_ + d] > 0.1f;

    float s = 0.f, s2 = 0.f, cnt = 0.f;
    for (int p = threadIdx.x; p < P_; p += 256) {
        if (!mask[b * P_ + p]) {
            float xv = x[((size_t)b * P_ + p) * D_ + d];
            float v  = isftr ? (ff ? xv : 0.f) : (ff ? 0.f : xv);
            s += v; s2 += v * v; cnt += 1.f;
        }
    }
    __shared__ float sh_s[256], sh_s2[256], sh_c[256];
    sh_s[threadIdx.x] = s; sh_s2[threadIdx.x] = s2; sh_c[threadIdx.x] = cnt;
    __syncthreads();
    for (int off = 128; off > 0; off >>= 1) {
        if ((int)threadIdx.x < off) {
            sh_s [threadIdx.x] += sh_s [threadIdx.x + off];
            sh_s2[threadIdx.x] += sh_s2[threadIdx.x + off];
            sh_c [threadIdx.x] += sh_c [threadIdx.x + off];
        }
        __syncthreads();
    }
    if (threadIdx.x == 0) {
        float n   = fmaxf(sh_c[0], 1.f);
        float mu  = sh_s[0] / n;
        float var = fmaxf(sh_s2[0] / n - mu * mu, 0.f);
        meanv[bc] = mu;
        stdv[bc]  = sqrtf(var);
    }
}

// ---------------------------------------------------------------------------
// Kernel 2: per point write xn (32 normalized+clipped channels), raw crd
// channels, and squared norm of crd part. One thread per point.
// ---------------------------------------------------------------------------
__global__ __launch_bounds__(256) void k_norm(const float* __restrict__ x,
                                              const float* __restrict__ features,
                                              const float* __restrict__ meanv,
                                              const float* __restrict__ stdv,
                                              float* __restrict__ xn,
                                              float* __restrict__ xcrd,
                                              float* __restrict__ sq)
{
    int pt = blockIdx.x * 256 + threadIdx.x;     // 0..32767
    int b  = pt >> 11;
    float s = 0.f;
    #pragma unroll
    for (int d = 0; d < D_; ++d) {
        float xv = x[(size_t)pt * D_ + d];
        bool  ff = features[b * D_ + d] > 0.1f;
        float c  = ff ? 0.f : xv;                // crd channel
        float f  = ff ? xv : 0.f;                // ftr channel
        xcrd[(size_t)pt * D_ + d] = c;
        s += c * c;
        float mu0 = meanv[b * 32 + d],      sd0 = stdv[b * 32 + d];
        float mu1 = meanv[b * 32 + 16 + d], sd1 = stdv[b * 32 + 16 + d];
        float n0 = fminf(fmaxf((c - mu0) / (sd0 + EPS_), -10.f), 10.f);
        float n1 = fminf(fmaxf((f - mu1) / (sd1 + EPS_), -10.f), 10.f);
        xn[(size_t)pt * 32 + d]      = n0;
        xn[(size_t)pt * 32 + 16 + d] = n1;
    }
    sq[pt] = s;
}

// ---------------------------------------------------------------------------
// Kernel 3: KNN. Block = 256 points of one batch; LDS-tiled candidate sweep.
// Ranks on SQUARED distance (sqrt is monotonic; only indices are consumed),
// saving one transcendental per candidate. Register-resident sorted top-16
// insertion, fully unrolled (ties keep the lower index = top_k stability).
// Prefetches the next candidate tile (global_prefetch_b8).
// ---------------------------------------------------------------------------
__global__ __launch_bounds__(256) void k_knn(const float* __restrict__ xcrd,
                                             const float* __restrict__ sq,
                                             const unsigned char* __restrict__ mask,
                                             int* __restrict__ knn_idx)
{
    __shared__ float         s_x[256 * D_];
    __shared__ float         s_sq[256];
    __shared__ unsigned char s_m[256];

    int b  = blockIdx.x >> 3;
    int p  = ((blockIdx.x & 7) << 8) + threadIdx.x;
    int pt = (b << 11) + p;

    float px[D_];
    #pragma unroll
    for (int d = 0; d < D_; ++d) px[d] = xcrd[(size_t)pt * D_ + d];
    float psq = sq[pt];

    float kd[KNN_]; int ki[KNN_];
    #pragma unroll
    for (int j = 0; j < KNN_; ++j) { kd[j] = __builtin_inff(); ki[j] = 0; }

    for (int qt = 0; qt < 8; ++qt) {
        int q0 = qt << 8;
        int qg = (b << 11) + q0 + threadIdx.x;
        #pragma unroll
        for (int d = 0; d < D_; ++d) s_x[threadIdx.x * D_ + d] = xcrd[(size_t)qg * D_ + d];
        s_sq[threadIdx.x] = sq[qg];
        s_m[threadIdx.x]  = mask[qg];
        if (qt < 7) {                         // prefetch next candidate tile
            __builtin_prefetch(&xcrd[(size_t)(qg + 256) * D_], 0, 3);
        }
        __syncthreads();

        for (int qq = 0; qq < 256; ++qq) {
            if (s_m[qq]) continue;             // masked -> dist = inf, never kept
            float dot = 0.f;
            #pragma unroll
            for (int d = 0; d < D_; ++d) dot = fmaf(px[d], s_x[qq * D_ + d], dot);
            float d2 = fmaxf(psq + s_sq[qq] - 2.f * dot, 0.f);   // rank on d^2
            if (d2 < kd[KNN_ - 1]) {
                int q = q0 + qq;
                #pragma unroll
                for (int j = KNN_ - 1; j >= 1; --j) {
                    bool sh  = d2 < kd[j - 1];
                    bool ins = (!sh) && (d2 < kd[j]);
                    kd[j] = sh ? kd[j - 1] : (ins ? d2 : kd[j]);
                    ki[j] = sh ? ki[j - 1] : (ins ? q  : ki[j]);
                }
                if (d2 < kd[0]) { kd[0] = d2; ki[0] = q; }
            }
        }
        __syncthreads();
    }
    #pragma unroll
    for (int j = 0; j < KNN_; ++j) knn_idx[(size_t)pt * KNN_ + j] = ki[j];
}

// ---------------------------------------------------------------------------
// Kernel 4: W (f32, COUT_ x CIN_, row-major) -> f16
// ---------------------------------------------------------------------------
__global__ __launch_bounds__(256) void k_wconv(const float* __restrict__ W,
                                               _Float16* __restrict__ Wh, int n)
{
    int i = blockIdx.x * 256 + threadIdx.x;
    if (i < n) Wh[i] = (_Float16)W[i];
}

// ---------------------------------------------------------------------------
// Kernel 5: build A (NPTS_ x 512, f16): [center | xn[i1]-center | ... ].
// One wave per point, lane = channel (0..31).
// ---------------------------------------------------------------------------
__global__ __launch_bounds__(256) void k_build_a(const float* __restrict__ xn,
                                                 const int* __restrict__ knn_idx,
                                                 _Float16* __restrict__ A)
{
    int pt = blockIdx.x * 8 + (threadIdx.x >> 5);
    int c  = threadIdx.x & 31;
    int base = (pt >> 11) << 11;                 // batch row base
    int q0 = knn_idx[(size_t)pt * KNN_];
    float center = xn[(size_t)(base + q0) * 32 + c];
    A[(size_t)pt * CIN_ + c] = (_Float16)center;
    #pragma unroll
    for (int j = 1; j < KNN_; ++j) {
        int qj = knn_idx[(size_t)pt * KNN_ + j];
        float v = xn[(size_t)(base + qj) * 32 + c] - center;
        A[(size_t)pt * CIN_ + j * 32 + c] = (_Float16)v;
    }
}

// ---------------------------------------------------------------------------
// Kernel 6: WMMA GEMM + fused bias + GLU epilogue.
// One wave computes TWO 16-row M-tiles (32 rows) for one 16-col tile of 'a'
// AND the matching tile of 'g' (columns col and col+512 of h): B fragments
// are loaded once and reused across both M-tiles (halves W traffic, 2 loads
// per WMMA). Then writes a*sigmoid(g).
// A fragment per ISA 16-bit A 16x32 layout; B fragment per 16-bit B layout
// (lanes 0-15: K 0-7,16-23, lanes 16-31: K 8-15,24-31; column = lane&15).
// ---------------------------------------------------------------------------
__global__ __launch_bounds__(256) void k_gemm(const _Float16* __restrict__ A,
                                              const _Float16* __restrict__ Wh,
                                              const float* __restrict__ bias,
                                              float* __restrict__ out)
{
    int wid  = (blockIdx.x * 256 + threadIdx.x) >> 5;
    int lane = threadIdx.x & 31;
    int mt = wid >> 5;                  // 0..1023 (32-row super-tiles)
    int nt = wid & 31;                  // 0..31
    int m0 = mt << 5;
    int n0 = nt << 4;
    int l15 = lane & 15;
    int hi  = lane >> 4;

    const _Float16* Ar0 = A + (size_t)(m0 + l15) * CIN_;
    const _Float16* Ar1 = Ar0 + (size_t)16 * CIN_;
    int col = n0 + l15;                            // output column (0..511)
    const _Float16* Wa = Wh + (size_t)col * CIN_;
    const _Float16* Wg = Wh + (size_t)(col + DM_) * CIN_;

    v8f acc_a0 = {}, acc_g0 = {}, acc_a1 = {}, acc_g1 = {};
    for (int k0 = 0; k0 < CIN_; k0 += 32) {
        int ka = k0 + hi * 8;          // A: lanes<16 K{0-7,16-23}, lanes>=16 K{8-15,24-31}
        v8h a0lo = *(const v8h*)(Ar0 + ka);
        v8h a0hi = *(const v8h*)(Ar0 + ka + 16);
        v8h a1lo = *(const v8h*)(Ar1 + ka);
        v8h a1hi = *(const v8h*)(Ar1 + ka + 16);
        int kb = k0 + hi * 8;          // B: lanes<16 K{0-7,16-23}, lanes>=16 K{8-15,24-31}
        v8h ba0 = *(const v8h*)(Wa + kb);
        v8h ba1 = *(const v8h*)(Wa + kb + 16);
        v8h bg0 = *(const v8h*)(Wg + kb);
        v8h bg1 = *(const v8h*)(Wg + kb + 16);
        v16h af0, af1, bfa, bfg;
        #pragma unroll
        for (int i = 0; i < 8; ++i) {
            af0[i] = a0lo[i]; af0[i + 8] = a0hi[i];
            af1[i] = a1lo[i]; af1[i + 8] = a1hi[i];
            bfa[i] = ba0[i];  bfa[i + 8] = ba1[i];
            bfg[i] = bg0[i];  bfg[i + 8] = bg1[i];
        }
        acc_a0 = __builtin_amdgcn_wmma_f32_16x16x32_f16(false, af0, false, bfa,
                                                        (short)0, acc_a0, false, false);
        acc_g0 = __builtin_amdgcn_wmma_f32_16x16x32_f16(false, af0, false, bfg,
                                                        (short)0, acc_g0, false, false);
        acc_a1 = __builtin_amdgcn_wmma_f32_16x16x32_f16(false, af1, false, bfa,
                                                        (short)0, acc_a1, false, false);
        acc_g1 = __builtin_amdgcn_wmma_f32_16x16x32_f16(false, af1, false, bfg,
                                                        (short)0, acc_g1, false, false);
    }

    float ba = bias[col];
    float bg = bias[col + DM_];
    int rbase = m0 + hi * 8;           // D layout: VGPR r -> row m0 + r (+8 for lanes>=16)
    #pragma unroll
    for (int r = 0; r < 8; ++r) {
        float av0 = acc_a0[r] + ba;
        float gv0 = acc_g0[r] + bg;
        out[(size_t)(rbase + r) * DM_ + col] = av0 / (1.0f + __expf(-gv0));
        float av1 = acc_a1[r] + ba;
        float gv1 = acc_g1[r] + bg;
        out[(size_t)(rbase + 16 + r) * DM_ + col] = av1 / (1.0f + __expf(-gv1));
    }
}

// ---------------------------------------------------------------------------
// Launch
// ---------------------------------------------------------------------------
extern "C" void kernel_launch(void* const* d_in, const int* in_sizes, int n_in,
                              void* d_out, int out_size, void* d_ws, size_t ws_size,
                              hipStream_t stream) {
    const float*         x        = (const float*)d_in[0];
    const float*         features = (const float*)d_in[1];
    const unsigned char* mask     = (const unsigned char*)d_in[2]; // all-false bool mask
    const float*         W        = (const float*)d_in[3];
    const float*         bias     = (const float*)d_in[4];
    float*               out      = (float*)d_out;

    char*  ws  = (char*)d_ws;
    size_t off = 0;
    auto alloc = [&](size_t bytes) {
        size_t o = off;
        off = (off + bytes + 255) & ~(size_t)255;
        return o;
    };
    float*    meanv   = (float*)   (ws + alloc((size_t)B_ * 32 * 4));
    float*    stdv    = (float*)   (ws + alloc((size_t)B_ * 32 * 4));
    float*    sq      = (float*)   (ws + alloc((size_t)NPTS_ * 4));
    float*    xcrd    = (float*)   (ws + alloc((size_t)NPTS_ * D_ * 4));
    float*    xn      = (float*)   (ws + alloc((size_t)NPTS_ * 32 * 4));
    int*      knn_idx = (int*)     (ws + alloc((size_t)NPTS_ * KNN_ * 4));
    _Float16* Wh      = (_Float16*)(ws + alloc((size_t)COUT_ * CIN_ * 2));
    _Float16* Amat    = (_Float16*)(ws + alloc((size_t)NPTS_ * CIN_ * 2));
    (void)ws_size; (void)in_sizes; (void)n_in; (void)out_size;   // ~43 MB of ws used

    k_stats  <<<B_ * 32,            256, 0, stream>>>(x, features, mask, meanv, stdv);
    k_norm   <<<NPTS_ / 256,        256, 0, stream>>>(x, features, meanv, stdv, xn, xcrd, sq);
    k_knn    <<<B_ * (P_ / 256),    256, 0, stream>>>(xcrd, sq, mask, knn_idx);
    k_wconv  <<<(COUT_ * CIN_) / 256, 256, 0, stream>>>(W, Wh, COUT_ * CIN_);
    k_build_a<<<NPTS_ / 8,          256, 0, stream>>>(xn, knn_idx, Amat);
    // waves = (NPTS_/32) * (DM_/16) = 32768 ; 8 waves per 256-thread block
    k_gemm   <<<(NPTS_ / 32) * (DM_ / 16) / 8, 256, 0, stream>>>(Amat, Wh, bias, out);
}
